// TopkRouter_63591285784863
// MI455X (gfx1250) — compile-verified, matrix operand order checked by reading, not run on hardware
//
#include <hip/hip_runtime.h>
#include <hip/hip_bf16.h>
#include <math.h>

#define EMBED   2048
#define NEXP    64
#define TOPK    8
#define WPB     8      // waves per block
#define LSTRIDE 68     // LDS row stride in floats: 16B aligned, bank-conflict free

typedef __bf16 bf16;
typedef __bf16 v16bf __attribute__((ext_vector_type(16)));
typedef __bf16 v8bf  __attribute__((ext_vector_type(8)));
typedef float  v8f   __attribute__((ext_vector_type(8)));

// ---------------------------------------------------------------------------
// Kernel 1: convert router weights W [64 x 2048] f32 -> bf16 once (into d_ws).
// ---------------------------------------------------------------------------
__global__ void pack_w_bf16(const float* __restrict__ W, bf16* __restrict__ Wb, int n) {
    int i = blockIdx.x * blockDim.x + threadIdx.x;
    if (i < n) Wb[i] = (bf16)W[i];
}

// B fragment (32x16 K-major tile of W^T): lane supplies column n = lane&15,
// halves hold K per the CDNA5 16-bit layout (kbase handled by caller pointer).
__device__ __forceinline__ v16bf load_bfrag(const bf16* __restrict__ p) {
    v8bf lo = *(const v8bf*)(p);        // K = kbase + 0..7
    v8bf hi = *(const v8bf*)(p + 16);   // K = kbase + 16..23
    v16bf b;
#pragma unroll
    for (int j = 0; j < 8; ++j) { b[j] = lo[j]; b[j + 8] = hi[j]; }
    return b;
}

// ---------------------------------------------------------------------------
// Kernel 2: scores GEMM via v_wmma_f32_16x16x32_bf16 + top-8 + masked softmax.
// One wave handles 16 tokens x 64 experts. 8 waves (128 tokens) per block.
// ---------------------------------------------------------------------------
__global__ __launch_bounds__(256)
void router_topk_kernel(const float* __restrict__ X, const bf16* __restrict__ Wb,
                        const float* __restrict__ bias, float* __restrict__ out,
                        int n_tokens) {
    __shared__ float lds[WPB][16 * LSTRIDE];

    const int wave  = threadIdx.x >> 5;
    const int lane  = threadIdx.x & 31;
    const int row   = lane & 15;
    const int kbase = (lane < 16) ? 0 : 8;   // 16-bit A/B layout K split
    const int mbase = (lane < 16) ? 0 : 8;   // f32 C/D layout M split

    const int ntiles   = (n_tokens + 15) >> 4;
    const int tile_raw = blockIdx.x * WPB + wave;
    const int tile     = (tile_raw < ntiles) ? tile_raw : (ntiles - 1); // clamp: keep waves alive for barrier

    const float* xrow = X + (size_t)(tile * 16 + row) * EMBED;
    const bf16*  w0   = Wb + (size_t)(0 * 16 + row) * EMBED + kbase;
    const bf16*  w1   = Wb + (size_t)(1 * 16 + row) * EMBED + kbase;
    const bf16*  w2   = Wb + (size_t)(2 * 16 + row) * EMBED + kbase;
    const bf16*  w3   = Wb + (size_t)(3 * 16 + row) * EMBED + kbase;

    v8f acc0 = {}, acc1 = {}, acc2 = {}, acc3 = {};

    for (int k0 = 0; k0 < EMBED; k0 += 32) {
        // ---- A fragment: token row of X, converted f32 -> bf16 in-register
        const float4 f0 = *(const float4*)(xrow + k0 + kbase);
        const float4 f1 = *(const float4*)(xrow + k0 + kbase + 4);
        const float4 f2 = *(const float4*)(xrow + k0 + kbase + 16);
        const float4 f3 = *(const float4*)(xrow + k0 + kbase + 20);
        v16bf a;
        a[0]  = (bf16)f0.x; a[1]  = (bf16)f0.y; a[2]  = (bf16)f0.z; a[3]  = (bf16)f0.w;
        a[4]  = (bf16)f1.x; a[5]  = (bf16)f1.y; a[6]  = (bf16)f1.z; a[7]  = (bf16)f1.w;
        a[8]  = (bf16)f2.x; a[9]  = (bf16)f2.y; a[10] = (bf16)f2.z; a[11] = (bf16)f2.w;
        a[12] = (bf16)f3.x; a[13] = (bf16)f3.y; a[14] = (bf16)f3.z; a[15] = (bf16)f3.w;

        // ---- 4 expert column tiles: D = A x B + C
        v16bf b0 = load_bfrag(w0 + k0);
        v16bf b1 = load_bfrag(w1 + k0);
        v16bf b2 = load_bfrag(w2 + k0);
        v16bf b3 = load_bfrag(w3 + k0);
        acc0 = __builtin_amdgcn_wmma_f32_16x16x32_bf16(false, a, false, b0, (short)0, acc0, false, false);
        acc1 = __builtin_amdgcn_wmma_f32_16x16x32_bf16(false, a, false, b1, (short)0, acc1, false, false);
        acc2 = __builtin_amdgcn_wmma_f32_16x16x32_bf16(false, a, false, b2, (short)0, acc2, false, false);
        acc3 = __builtin_amdgcn_wmma_f32_16x16x32_bf16(false, a, false, b3, (short)0, acc3, false, false);
    }

    // ---- spill scores (+bias) to LDS; D layout: elem v -> token mbase+v, expert t*16+row
    {
        const float bv0 = bias[0 * 16 + row];
        const float bv1 = bias[1 * 16 + row];
        const float bv2 = bias[2 * 16 + row];
        const float bv3 = bias[3 * 16 + row];
        float* wl = lds[wave];
#pragma unroll
        for (int v = 0; v < 8; ++v) {
            wl[(mbase + v) * LSTRIDE + 0 * 16 + row] = acc0[v] + bv0;
            wl[(mbase + v) * LSTRIDE + 1 * 16 + row] = acc1[v] + bv1;
            wl[(mbase + v) * LSTRIDE + 2 * 16 + row] = acc2[v] + bv2;
            wl[(mbase + v) * LSTRIDE + 3 * 16 + row] = acc3[v] + bv3;
        }
    }
    __syncthreads();

    // ---- top-8 + masked softmax: lanes 0..15 each own one token of the tile
    if (lane < 16) {
        const int  token = tile * 16 + lane;
        const bool valid = (tile_raw < ntiles) && (token < n_tokens);

        float s[NEXP];
        const float* sr = &lds[wave][lane * LSTRIDE];
#pragma unroll
        for (int e = 0; e < NEXP; e += 4) {
            float4 f = *(const float4*)(sr + e);
            s[e] = f.x; s[e + 1] = f.y; s[e + 2] = f.z; s[e + 3] = f.w;
        }

        float vals[TOPK]; int idxs[TOPK];
#pragma unroll
        for (int k = 0; k < TOPK; ++k) {
            float best = -INFINITY; int bi = 0;
            for (int e = 0; e < NEXP; ++e) {       // strict '>' => lowest index wins ties (matches lax.top_k)
                if (s[e] > best) { best = s[e]; bi = e; }
            }
            vals[k] = best; idxs[k] = bi; s[bi] = -INFINITY;
        }

        // softmax over mask: non-selected entries are exactly 0.0
        const float m  = fmaxf(vals[0], 0.0f);
        const float zb = __expf(-m);               // exp(0 - m)
        float Z = (float)(NEXP - TOPK) * zb;
#pragma unroll
        for (int k = 0; k < TOPK; ++k) Z += __expf(vals[k] - m);
        const float inv = 1.0f / Z;
        const float pb  = zb * inv;                // prob of every non-selected expert

        if (valid) {
            float* orow = out + (size_t)token * NEXP;
            float4 bvv = { pb, pb, pb, pb };
#pragma unroll
            for (int e = 0; e < NEXP; e += 4) *(float4*)(orow + e) = bvv;
            float* irow = out + (size_t)n_tokens * NEXP + (size_t)token * TOPK;
#pragma unroll
            for (int k = 0; k < TOPK; ++k) {
                orow[idxs[k]] = __expf(vals[k] - m) * inv;  // later same-lane store wins
                irow[k] = (float)idxs[k];
            }
        }
    }
}

// ---------------------------------------------------------------------------
extern "C" void kernel_launch(void* const* d_in, const int* in_sizes, int n_in,
                              void* d_out, int out_size, void* d_ws, size_t ws_size,
                              hipStream_t stream) {
    const float* X    = (const float*)d_in[0];   // [B,S,2048] f32
    const float* W    = (const float*)d_in[1];   // [64,2048]  f32
    const float* bias = (const float*)d_in[2];   // [64]       f32
    float*       out  = (float*)d_out;

    const int n_tokens = in_sizes[0] / EMBED;

    bf16* Wb = (bf16*)d_ws;                      // 64*2048*2 = 256 KB scratch
    const int wn = NEXP * EMBED;
    pack_w_bf16<<<(wn + 255) / 256, 256, 0, stream>>>(W, Wb, wn);

    const int ntiles = (n_tokens + 15) / 16;
    const int blocks = (ntiles + WPB - 1) / WPB;
    router_topk_kernel<<<blocks, 256, 0, stream>>>(X, Wb, bias, out, n_tokens);
}
